// ChemicalNet_85435489452601
// MI455X (gfx1250) — compile-verified, matrix operand order checked by reading, not run on hardware
//
#include <hip/hip_runtime.h>
#include <hip/hip_bf16.h>

#define NATOMS_MAX 150000
#define NSPEC 8
#define DIN 256
#define H1 256
#define H2 256
#define H3 128
#define MT 32            // atoms per block tile (two 16-row WMMA M-tiles)
#define LDP (DIN + 8)    // padded LDS row stride (528B) -> no 64-bank conflicts

typedef __attribute__((ext_vector_type(16))) __bf16 bf16x16;
typedef __attribute__((ext_vector_type(8)))  __bf16 bf16x8;
typedef __attribute__((ext_vector_type(8)))  float  f32x8;

union Frag16 { bf16x16 v; bf16x8 h[2]; };

__device__ __forceinline__ __bf16 f2bf(float f) {
    union { float f; unsigned u; } in; in.f = f;
    unsigned u = in.u;
    unsigned r = (u + 0x7FFFu + ((u >> 16) & 1u)) >> 16;   // RNE
    union { unsigned short s; __bf16 b; } out; out.s = (unsigned short)r;
    return out.b;
}

__device__ __forceinline__ int spec_slot(int z) {
    switch (z) {
        case 1:  return 0; case 6:  return 1; case 7:  return 2; case 8:  return 3;
        case 9:  return 4; case 15: return 5; case 16: return 6; case 17: return 7;
    }
    return -1;
}

__device__ __forceinline__ float silu(float x) {
    return x * (1.0f / (1.0f + __expf(-x)));
}

// ---------------- bucketing kernels ----------------

__global__ void init_k(int* counts) {
    if (threadIdx.x < 24) counts[threadIdx.x] = 0;   // counts[8], cursors[8], offsets[8]
}

__global__ void count_k(const int* __restrict__ species, int n, int* __restrict__ counts) {
    int i = blockIdx.x * blockDim.x + threadIdx.x;
    if (i < n) {
        int s = spec_slot(species[i]);
        if (s >= 0) atomicAdd(&counts[s], 1);
    }
}

__global__ void scan_k(const int* __restrict__ counts, int* __restrict__ offsets) {
    if (blockIdx.x == 0 && threadIdx.x == 0) {
        int acc = 0;
        for (int i = 0; i < NSPEC; ++i) { offsets[i] = acc; acc += counts[i]; }
    }
}

__global__ void scatter_k(const int* __restrict__ species, int n,
                          const int* __restrict__ offsets, int* __restrict__ cursors,
                          int* __restrict__ perm) {
    int i = blockIdx.x * blockDim.x + threadIdx.x;
    if (i < n) {
        int s = spec_slot(species[i]);
        if (s >= 0) {
            int pos = atomicAdd(&cursors[s], 1);
            perm[offsets[s] + pos] = i;
        }
    }
}

// fp32 W[s][d][h]  ->  bf16 Wt[s][h][d]   (row-major in d = K, so B-frags load contiguously)
__global__ void wconv_k(const float* __restrict__ W, __bf16* __restrict__ Wt, int D, int H) {
    int s = blockIdx.y;
    const float* Ws = W + (size_t)s * D * H;
    __bf16* Wts = Wt + (size_t)s * D * H;
    for (int e = blockIdx.x * blockDim.x + threadIdx.x; e < D * H; e += gridDim.x * blockDim.x) {
        int d = e / H, h = e % H;
        Wts[(size_t)h * D + d] = f2bf(Ws[e]);
    }
}

// ---------------- WMMA MLP kernel ----------------

// Two 16x16 output tiles (M-rows 0..15 and 16..31) sharing each B fragment:
// halves L2 weight traffic vs one-tile-at-a-time.
__device__ __forceinline__ void tile_mm2(const __bf16 (*__restrict__ src)[LDP],
                                         const __bf16* __restrict__ wrow,
                                         int laneLo, int laneHi,
                                         f32x8& acc0, f32x8& acc1) {
#pragma unroll
    for (int ks = 0; ks < DIN / 32; ++ks) {
        const int k0 = ks * 32 + laneHi * 8;
        Frag16 b, a0, a1;
        b.h[0]  = *(const bf16x8*)(wrow + k0);
        b.h[1]  = *(const bf16x8*)(wrow + k0 + 16);
        a0.h[0] = *(const bf16x8*)&src[laneLo][k0];
        a0.h[1] = *(const bf16x8*)&src[laneLo][k0 + 16];
        a1.h[0] = *(const bf16x8*)&src[16 + laneLo][k0];
        a1.h[1] = *(const bf16x8*)&src[16 + laneLo][k0 + 16];
        acc0 = __builtin_amdgcn_wmma_f32_16x16x32_bf16(false, a0.v, false, b.v,
                                                       (short)0, acc0, false, false);
        acc1 = __builtin_amdgcn_wmma_f32_16x16x32_bf16(false, a1.v, false, b.v,
                                                       (short)0, acc1, false, false);
    }
}

__global__ __launch_bounds__(256) void mlp_k(
    const float* __restrict__ emb,
    const float* __restrict__ b1, const float* __restrict__ b2, const float* __restrict__ b3,
    const int* __restrict__ counts, const int* __restrict__ offsets,
    const int* __restrict__ perm,
    const __bf16* __restrict__ Wt1, const __bf16* __restrict__ Wt2,
    const __bf16* __restrict__ Wt3,
    float* __restrict__ out) {

    __shared__ __bf16 buf0[MT][LDP];
    __shared__ __bf16 buf1[MT][LDP];
    __shared__ int aidx[MT];

    const int s = blockIdx.y;
    const int cnt = counts[s];
    const int m0 = blockIdx.x * MT;
    if (m0 >= cnt) return;                 // uniform per block
    const int valid = (cnt - m0 < MT) ? (cnt - m0) : MT;
    const int base = offsets[s] + m0;
    const int tid = threadIdx.x;

    if (tid < MT) {
        int j = (tid < valid) ? tid : (valid - 1);   // pad with last valid atom
        aidx[tid] = perm[base + j];
    }
    __syncthreads();

    // stage embeddings fp32 -> bf16; float4 (b128) loads, 256B per 16 lanes, 64 lanes per row
    for (int e = tid; e < MT * (DIN / 4); e += 256) {
        int r = e >> 6, c4 = (e & 63) * 4;
        const float4 v = *(const float4*)(emb + (size_t)aidx[r] * DIN + c4);
        __bf16* dst = &buf0[r][c4];
        dst[0] = f2bf(v.x); dst[1] = f2bf(v.y); dst[2] = f2bf(v.z); dst[3] = f2bf(v.w);
    }
    __syncthreads();

    const int wid = tid >> 5;
    const int lane = tid & 31;
    const int laneLo = lane & 15;
    const int laneHi = lane >> 4;

    const __bf16* Wt1s = Wt1 + (size_t)s * H1 * DIN;
    const __bf16* Wt2s = Wt2 + (size_t)s * H2 * H1;
    const __bf16* Wt3s = Wt3 + (size_t)s * H3 * H2;
    const float* b1s = b1 + s * H1;
    const float* b2s = b2 + s * H2;
    const float* b3s = b3 + s * H3;

    // ---- layer 1: buf0 x Wt1 -> silu -> buf1 ----
    for (int nt = wid; nt < H1 / 16; nt += 8) {
        const int n = nt * 16 + laneLo;
        if (nt + 8 < H1 / 16)
            __builtin_prefetch(Wt1s + (size_t)((nt + 8) * 16 + laneLo) * DIN, 0, 0);
        f32x8 acc0 = {0.f,0.f,0.f,0.f,0.f,0.f,0.f,0.f};
        f32x8 acc1 = {0.f,0.f,0.f,0.f,0.f,0.f,0.f,0.f};
        tile_mm2(buf0, Wt1s + (size_t)n * DIN, laneLo, laneHi, acc0, acc1);
        const float bb = b1s[n];
#pragma unroll
        for (int r = 0; r < 8; ++r) {
            const int m = laneHi * 8 + r;
            float x0 = acc0[r] + bb;
            float x1 = acc1[r] + bb;
            buf1[m][n]      = f2bf(silu(x0));
            buf1[m + 16][n] = f2bf(silu(x1));
        }
    }
    __syncthreads();

    // ---- layer 2: buf1 x Wt2 -> silu -> buf0 ----
    for (int nt = wid; nt < H2 / 16; nt += 8) {
        const int n = nt * 16 + laneLo;
        if (nt + 8 < H2 / 16)
            __builtin_prefetch(Wt2s + (size_t)((nt + 8) * 16 + laneLo) * H1, 0, 0);
        f32x8 acc0 = {0.f,0.f,0.f,0.f,0.f,0.f,0.f,0.f};
        f32x8 acc1 = {0.f,0.f,0.f,0.f,0.f,0.f,0.f,0.f};
        tile_mm2(buf1, Wt2s + (size_t)n * H1, laneLo, laneHi, acc0, acc1);
        const float bb = b2s[n];
#pragma unroll
        for (int r = 0; r < 8; ++r) {
            const int m = laneHi * 8 + r;
            float x0 = acc0[r] + bb;
            float x1 = acc1[r] + bb;
            buf0[m][n]      = f2bf(silu(x0));
            buf0[m + 16][n] = f2bf(silu(x1));
        }
    }
    __syncthreads();

    // ---- layer 3: buf0 x Wt3 -> +bias -> global out (fp32) ----
    for (int nt = wid; nt < H3 / 16; nt += 8) {
        const int n = nt * 16 + laneLo;
        f32x8 acc0 = {0.f,0.f,0.f,0.f,0.f,0.f,0.f,0.f};
        f32x8 acc1 = {0.f,0.f,0.f,0.f,0.f,0.f,0.f,0.f};
        tile_mm2(buf0, Wt3s + (size_t)n * H2, laneLo, laneHi, acc0, acc1);
        const float bb = b3s[n];
#pragma unroll
        for (int r = 0; r < 8; ++r) {
            const int m = laneHi * 8 + r;
            if (m < valid)      out[(size_t)aidx[m]      * H3 + n] = acc0[r] + bb;
            if (m + 16 < valid) out[(size_t)aidx[m + 16] * H3 + n] = acc1[r] + bb;
        }
    }
}

// ---------------- host launcher ----------------

extern "C" void kernel_launch(void* const* d_in, const int* in_sizes, int n_in,
                              void* d_out, int out_size, void* d_ws, size_t ws_size,
                              hipStream_t stream) {
    const int*   species = (const int*)  d_in[0];
    const float* emb     = (const float*)d_in[1];
    const float* W1      = (const float*)d_in[2];
    const float* b1      = (const float*)d_in[3];
    const float* W2      = (const float*)d_in[4];
    const float* b2      = (const float*)d_in[5];
    const float* W3      = (const float*)d_in[6];
    const float* b3      = (const float*)d_in[7];
    const int n = in_sizes[0];

    char* ws = (char*)d_ws;
    int* counts  = (int*)ws;          // 8
    int* cursors = counts + 8;        // 8
    int* offsets = cursors + 8;       // 8
    int* perm    = (int*)(ws + 128);
    size_t o = (128 + (size_t)n * 4 + 255) & ~(size_t)255;
    __bf16* Wt1 = (__bf16*)(ws + o);                          o += (size_t)NSPEC * H1 * DIN * 2;
    __bf16* Wt2 = (__bf16*)(ws + o);                          o += (size_t)NSPEC * H2 * H1 * 2;
    __bf16* Wt3 = (__bf16*)(ws + o);
    (void)ws_size; (void)n_in;

    hipMemsetAsync(d_out, 0, (size_t)out_size * sizeof(float), stream);
    init_k<<<1, 64, 0, stream>>>(counts);

    const int nb = (n + 255) / 256;
    count_k<<<nb, 256, 0, stream>>>(species, n, counts);
    scan_k<<<1, 32, 0, stream>>>(counts, offsets);
    scatter_k<<<nb, 256, 0, stream>>>(species, n, offsets, cursors, perm);

    wconv_k<<<dim3(64, NSPEC), 256, 0, stream>>>(W1, Wt1, DIN, H1);
    wconv_k<<<dim3(64, NSPEC), 256, 0, stream>>>(W2, Wt2, H1, H2);
    wconv_k<<<dim3(32, NSPEC), 256, 0, stream>>>(W3, Wt3, H2, H3);

    dim3 grid((n + MT - 1) / MT, NSPEC);
    mlp_k<<<grid, 256, 0, stream>>>(emb, b1, b2, b3, counts, offsets, perm,
                                    Wt1, Wt2, Wt3, (float*)d_out);
}
